// DecoderBlock_79302276153487
// MI455X (gfx1250) — compile-verified, matrix operand order checked by reading, not run on hardware
//
#include <hip/hip_runtime.h>
#include <stdint.h>

// ---- problem constants (hardcoded from reference) ----
#define Bb   64
#define Tt   512
#define Dd   384
#define Hh   6
#define HSs  64
#define MT   ((size_t)Bb * Tt)      // 32768 rows
#define Dh   (4 * Dd)               // 1536 MLP hidden

typedef float v8f  __attribute__((ext_vector_type(8)));
typedef __bf16 bf16_t;
typedef bf16_t v16bf __attribute__((ext_vector_type(16)));

// float -> bf16 bits, round-to-nearest-even
__device__ __forceinline__ uint16_t f2bf(float f) {
    union { float f; uint32_t u; } c; c.f = f;
    uint32_t r = c.u + 0x7FFFu + ((c.u >> 16) & 1u);
    return (uint16_t)(r >> 16);
}

// ---- DPP 16-lane (half-wave) butterfly reductions: pure VALU, no LDS ----
// DPP row = 16 lanes on wave32. Sequence xor1, xor2, half_mirror, mirror
// reduces across each 16-lane row; every lane ends with the row result.
template <int CTRL>
__device__ __forceinline__ float dpp_mov(float x) {
    int r = __builtin_amdgcn_update_dpp(0, __float_as_int(x), CTRL, 0xF, 0xF, true);
    return __int_as_float(r);
}
__device__ __forceinline__ float red16_max(float x) {
    x = fmaxf(x, dpp_mov<0xB1>(x));    // quad_perm [1,0,3,2]  (xor 1)
    x = fmaxf(x, dpp_mov<0x4E>(x));    // quad_perm [2,3,0,1]  (xor 2)
    x = fmaxf(x, dpp_mov<0x141>(x));   // row_half_mirror
    x = fmaxf(x, dpp_mov<0x140>(x));   // row_mirror
    return x;
}
__device__ __forceinline__ float red16_sum(float x) {
    x += dpp_mov<0xB1>(x);
    x += dpp_mov<0x4E>(x);
    x += dpp_mov<0x141>(x);
    x += dpp_mov<0x140>(x);
    return x;
}

// Load a 16x32 bf16 fragment (A-layout; also B^T rows for the B operand).
// base points at element [row0, k0] of a row-major matrix with leading dim ld.
// Per ISA: lanes 0-15 row M=lane, K in {0..7}U{16..23}; lanes 16-31 same rows,
// K in {8..15}U{24..31}; element pairs (2r,2r+1) are consecutive K -> u32 loads.
__device__ __forceinline__ v16bf load_frag(const uint16_t* base, int ld, int lane) {
    const uint16_t* p = base + (size_t)(lane & 15) * ld + ((lane & 16) ? 8 : 0);
    union { v16bf v; uint32_t u[8]; } f;
    const uint32_t* p0 = (const uint32_t*)p;         // K = khalf + 0..7
    const uint32_t* p1 = (const uint32_t*)(p + 16);  // K = khalf + 16..23
#pragma unroll
    for (int i = 0; i < 4; ++i) { f.u[i] = p0[i]; f.u[4 + i] = p1[i]; }
    return f.v;
}

__device__ __forceinline__ v8f wmma_bf16(v16bf a, v16bf b, v8f c) {
    return __builtin_amdgcn_wmma_f32_16x16x32_bf16(false, a, false, b, (short)0, c, false, false);
}

// ---------------- weight conversion ----------------
// Wq/Wk/Wv: (H, D, HS) f32 -> [n = h*HS+e][k] bf16 (transposed per head)
__global__ void conv_qkv_kernel(const float* __restrict__ W, uint16_t* __restrict__ out) {
    int i = blockIdx.x * blockDim.x + threadIdx.x;
    if (i >= Hh * Dd * HSs) return;
    int e = i % HSs;
    int k = (i / HSs) % Dd;
    int h = i / (HSs * Dd);
    out[(size_t)(h * HSs + e) * Dd + k] = f2bf(W[i]);
}

// W: (K, N) f32 -> out[n*K + k] bf16 (transpose)
__global__ void conv_t_kernel(const float* __restrict__ W, uint16_t* __restrict__ out,
                              int K, int N) {
    int i = blockIdx.x * blockDim.x + threadIdx.x;
    if (i >= K * N) return;
    int n = i % N, k = i / N;
    out[(size_t)n * K + k] = f2bf(W[i]);
}

// ---------------- LayerNorm (one wave per 384-elem row) ----------------
__global__ void ln_kernel(const float* __restrict__ x, const float* __restrict__ g,
                          const float* __restrict__ bta, uint16_t* __restrict__ out) {
    int wave = threadIdx.x >> 5, lane = threadIdx.x & 31;
    int row  = blockIdx.x * 8 + wave;
    const float* xr = x + (size_t)row * Dd;
    float v[12]; float s = 0.f;
#pragma unroll
    for (int i = 0; i < 12; ++i) { v[i] = xr[lane + 32 * i]; s += v[i]; }
#pragma unroll
    for (int m = 16; m >= 1; m >>= 1) s += __shfl_xor(s, m, 32);
    float mean = s * (1.0f / Dd);
    float s2 = 0.f;
#pragma unroll
    for (int i = 0; i < 12; ++i) { float d = v[i] - mean; s2 += d * d; }
#pragma unroll
    for (int m = 16; m >= 1; m >>= 1) s2 += __shfl_xor(s2, m, 32);
    float rstd = rsqrtf(s2 * (1.0f / Dd) + 1e-5f);
#pragma unroll
    for (int i = 0; i < 12; ++i) {
        int c = lane + 32 * i;
        out[(size_t)row * Dd + c] = f2bf((v[i] - mean) * rstd * g[c] + bta[c]);
    }
}

// ---------------- WMMA GEMM: C[M,N] = A[M,K](bf16) * Bt[N,K](bf16) ----------------
// epilogue: +bias[n] (opt), +resid[m,n] (opt), relu (tmpl), out f32 or bf16 (tmpl)
// one wave -> 32x64 tile (2 A-frags x 4 B-frags -> 8 WMMAs/k-step),
// 8 waves/block -> 256 rows/block
template <bool RELU, bool OUT_BF16>
__global__ void gemm_bf16_kernel(const uint16_t* __restrict__ A,
                                 const uint16_t* __restrict__ Bt,
                                 const float* __restrict__ bias,
                                 const float* __restrict__ resid,
                                 float* __restrict__ outF,
                                 uint16_t* __restrict__ outB,
                                 int M, int N, int K) {
    const int lane = threadIdx.x & 31;
    const int wave = threadIdx.x >> 5;
    const int m0 = (blockIdx.y * 8 + wave) * 32;
    const int n0 = blockIdx.x * 64;
    if (m0 >= M) return;

    v8f acc[2][4] = {};
    const uint16_t* Ap0 = A + (size_t)m0 * K;
    const uint16_t* Ap1 = A + (size_t)(m0 + 16) * K;
    const uint16_t* Bp[4];
#pragma unroll
    for (int j = 0; j < 4; ++j) Bp[j] = Bt + (size_t)(n0 + 16 * j) * K;

    for (int k0 = 0; k0 < K; k0 += 32) {
        v16bf a0 = load_frag(Ap0 + k0, K, lane);
        v16bf a1 = load_frag(Ap1 + k0, K, lane);
#pragma unroll
        for (int j = 0; j < 4; ++j) {
            v16bf b = load_frag(Bp[j] + k0, K, lane);
            acc[0][j] = wmma_bf16(a0, b, acc[0][j]);
            acc[1][j] = wmma_bf16(a1, b, acc[1][j]);
        }
    }

    const int col   = lane & 15;
    const int rbase = (lane >> 4) * 8;  // C-frag: M = r + 8*(lane>=16)
#pragma unroll
    for (int t = 0; t < 2; ++t) {
#pragma unroll
        for (int j = 0; j < 4; ++j) {
            int n = n0 + 16 * j + col;
            float bv = bias ? bias[n] : 0.0f;
#pragma unroll
            for (int r = 0; r < 8; ++r) {
                int m = m0 + 16 * t + rbase + r;
                float v = acc[t][j][r] + bv;
                if (resid) v += resid[(size_t)m * N + n];
                if (RELU) v = fmaxf(v, 0.0f);
                size_t oi = (size_t)m * N + n;
                if (OUT_BF16) outB[oi] = f2bf(v);
                else          outF[oi] = v;
            }
        }
    }
}

// ---------------- Flash attention (causal, scale = 1/sqrt(D)) ----------------
// q,k,v, out: [B, T, H*HS] bf16. Block = 4 waves; wave = 16 query rows.
// grid: (T/64, B*H)
__global__ void attn_kernel(const uint16_t* __restrict__ qbf,
                            const uint16_t* __restrict__ kbf,
                            const uint16_t* __restrict__ vbf,
                            uint16_t* __restrict__ attn_out) {
    __shared__ uint16_t ktile[32 * 64];      // [key][hs],  ld 64
    __shared__ uint16_t vT[64 * 32];         // [hs][key],  ld 32
    __shared__ uint16_t pst[4][16 * 32];     // per-wave P [row][keylocal], ld 32

    const int tid  = threadIdx.x;
    const int lane = tid & 31;
    const int wave = tid >> 5;
    const int hl   = lane >> 4;
    const int col  = lane & 15;

    const int bh = blockIdx.y;
    const int b  = bh / Hh, h = bh % Hh;
    const int q0b = blockIdx.x * 64;
    const int q0  = q0b + wave * 16;

    const size_t rowbase = (size_t)b * Tt * Dd + (size_t)h * HSs;  // + t*Dd

    // Q fragments (hs 0..31 and 32..63), resident for whole pass
    v16bf aq0 = load_frag(qbf + rowbase + (size_t)q0 * Dd,      Dd, lane);
    v16bf aq1 = load_frag(qbf + rowbase + (size_t)q0 * Dd + 32, Dd, lane);

    v8f   acc[4] = {};
    float mrun[8], lrun[8];
#pragma unroll
    for (int r = 0; r < 8; ++r) { mrun[r] = -1e30f; lrun[r] = 0.f; }

    const float scale = 0.051031036307982884f;  // 1/sqrt(384) (reference uses n_embd!)
    const int nkt = q0b / 32 + 2;               // key tiles of 32 needed by this block

    for (int kt = 0; kt < nkt; ++kt) {
        const int kbase = kt * 32;
        __syncthreads();
        // stage K tile [key][hs] as dwords (coalesced)
        {
            uint32_t* kd = (uint32_t*)ktile;
#pragma unroll
            for (int i = 0; i < 8; ++i) {
                int f = tid * 8 + i;                    // 0..1023 dwords
                int key = f >> 5, eu = f & 31;
                kd[f] = ((const uint32_t*)(kbf + rowbase + (size_t)(kbase + key) * Dd))[eu];
            }
            // stage V transposed [hs][key]
#pragma unroll
            for (int i = 0; i < 16; ++i) {
                int f = tid * 16 + i;                   // f = e*32 + key
                int e = f >> 5, key = f & 31;
                vT[f] = vbf[rowbase + (size_t)(kbase + key) * Dd + e];
            }
        }
        __syncthreads();

        const bool active = (kbase <= q0 + 15);
        if (active) {
            // scores: S[16 x 32] in two C-frags
            v8f s0 = {}, s1 = {};
            v16bf bk0a = load_frag(ktile,               64, lane);  // keys 0..15, hs 0..31
            v16bf bk0b = load_frag(ktile + 32,          64, lane);  // keys 0..15, hs 32..63
            v16bf bk1a = load_frag(ktile + 16 * 64,     64, lane);  // keys 16..31
            v16bf bk1b = load_frag(ktile + 16 * 64 + 32,64, lane);
            s0 = wmma_bf16(aq0, bk0a, s0); s0 = wmma_bf16(aq1, bk0b, s0);
            s1 = wmma_bf16(aq0, bk1a, s1); s1 = wmma_bf16(aq1, bk1b, s1);

            // online softmax per row; rows live in 16-lane halves -> DPP reductions
#pragma unroll
            for (int r = 0; r < 8; ++r) {
                int row = q0 + r + 8 * hl;
                int c0 = kbase + col, c1 = kbase + 16 + col;
                float x0 = (c0 <= row) ? s0[r] * scale : -1e30f;
                float x1 = (c1 <= row) ? s1[r] * scale : -1e30f;
                float mx = red16_max(fmaxf(x0, x1));
                float mnew = fmaxf(mrun[r], mx);
                float p0 = __expf(x0 - mnew);
                float p1 = __expf(x1 - mnew);
                float ps = red16_sum(p0 + p1);
                float alpha = __expf(mrun[r] - mnew);
                lrun[r] = lrun[r] * alpha + ps;
                mrun[r] = mnew;
#pragma unroll
                for (int j = 0; j < 4; ++j) acc[j][r] *= alpha;
                // stage P (C-frag layout -> [row][keylocal] in LDS)
                pst[wave][(r + 8 * hl) * 32 + col]      = f2bf(p0);
                pst[wave][(r + 8 * hl) * 32 + 16 + col] = f2bf(p1);
            }
        }
        __syncthreads();  // P visible; uniform barrier across the block
        if (active) {
            v16bf ap = load_frag(pst[wave], 32, lane);  // P as A-frag 16x32
#pragma unroll
            for (int j = 0; j < 4; ++j) {
                v16bf bv = load_frag(vT + (16 * j) * 32, 32, lane);  // V^T rows = hs
                acc[j] = wmma_bf16(ap, bv, acc[j]);
            }
        }
    }

    // normalize and write [B,T,H*HS]
#pragma unroll
    for (int j = 0; j < 4; ++j) {
#pragma unroll
        for (int r = 0; r < 8; ++r) {
            int row = q0 + r + 8 * hl;
            float v = acc[j][r] / lrun[r];
            attn_out[rowbase + (size_t)row * Dd + 16 * j + col] = f2bf(v);
        }
    }
}

// ---------------- launcher ----------------
extern "C" void kernel_launch(void* const* d_in, const int* in_sizes, int n_in,
                              void* d_out, int out_size, void* d_ws, size_t ws_size,
                              hipStream_t stream) {
    (void)in_sizes; (void)n_in; (void)out_size; (void)ws_size;
    const float* x   = (const float*)d_in[0];
    const float* Wq  = (const float*)d_in[1];
    const float* Wk  = (const float*)d_in[2];
    const float* Wv  = (const float*)d_in[3];
    const float* Wo  = (const float*)d_in[4];
    const float* bo  = (const float*)d_in[5];
    const float* g1  = (const float*)d_in[6];
    const float* b1  = (const float*)d_in[7];
    const float* g2  = (const float*)d_in[8];
    const float* b2  = (const float*)d_in[9];
    const float* W1  = (const float*)d_in[10];
    const float* bb1 = (const float*)d_in[11];
    const float* W2  = (const float*)d_in[12];
    const float* bb2 = (const float*)d_in[13];
    float* out = (float*)d_out;

    char* ws = (char*)d_ws;
    size_t off = 0;
    auto take = [&](size_t bytes) -> char* {
        char* p = ws + off;
        off += (bytes + 255) & ~(size_t)255;
        return p;
    };

    uint16_t* h_bf   = (uint16_t*)take(MT * Dd * 2);   // LN1 out; reused for LN2 out
    uint16_t* qbf    = (uint16_t*)take(MT * Dd * 2);
    uint16_t* kbf    = (uint16_t*)take(MT * Dd * 2);
    uint16_t* vbf    = (uint16_t*)take(MT * Dd * 2);
    uint16_t* attnbf = (uint16_t*)take(MT * Dd * 2);
    uint16_t* hidbf  = qbf;                            // reuse q/k/v/attn = MT*4D bf16
    float*    x1     = (float*)take(MT * Dd * 4);
    uint16_t* Wqt    = (uint16_t*)take((size_t)Dd * Dd * 2);
    uint16_t* Wkt    = (uint16_t*)take((size_t)Dd * Dd * 2);
    uint16_t* Wvt    = (uint16_t*)take((size_t)Dd * Dd * 2);
    uint16_t* Wot    = (uint16_t*)take((size_t)Dd * Dd * 2);
    uint16_t* W1t    = (uint16_t*)take((size_t)Dd * Dh * 2);
    uint16_t* W2t    = (uint16_t*)take((size_t)Dh * Dd * 2);

    // weight conversion (to transposed bf16)
    {
        int nq = Hh * Dd * HSs;
        conv_qkv_kernel<<<(nq + 255) / 256, 256, 0, stream>>>(Wq, Wqt);
        conv_qkv_kernel<<<(nq + 255) / 256, 256, 0, stream>>>(Wk, Wkt);
        conv_qkv_kernel<<<(nq + 255) / 256, 256, 0, stream>>>(Wv, Wvt);
        conv_t_kernel<<<(Dd * Dd + 255) / 256, 256, 0, stream>>>(Wo, Wot, Dd, Dd);
        conv_t_kernel<<<(Dd * Dh + 255) / 256, 256, 0, stream>>>(W1, W1t, Dd, Dh);
        conv_t_kernel<<<(Dh * Dd + 255) / 256, 256, 0, stream>>>(W2, W2t, Dh, Dd);
    }

    const dim3 blk(256);
    const dim3 gD(Dd / 64, (int)(MT / 256));   // N=384 GEMMs, 256 rows/block
    const dim3 gH(Dh / 64, (int)(MT / 256));   // N=1536 GEMM

    // LN1
    ln_kernel<<<(int)(MT / 8), blk, 0, stream>>>(x, g1, b1, h_bf);
    // QKV projections
    gemm_bf16_kernel<false, true><<<gD, blk, 0, stream>>>(h_bf, Wqt, nullptr, nullptr,
                                                          nullptr, qbf, (int)MT, Dd, Dd);
    gemm_bf16_kernel<false, true><<<gD, blk, 0, stream>>>(h_bf, Wkt, nullptr, nullptr,
                                                          nullptr, kbf, (int)MT, Dd, Dd);
    gemm_bf16_kernel<false, true><<<gD, blk, 0, stream>>>(h_bf, Wvt, nullptr, nullptr,
                                                          nullptr, vbf, (int)MT, Dd, Dd);
    // attention
    attn_kernel<<<dim3(Tt / 64, Bb * Hh), dim3(128), 0, stream>>>(qbf, kbf, vbf, attnbf);
    // output projection + residual -> x1 (f32)
    gemm_bf16_kernel<false, false><<<gD, blk, 0, stream>>>(attnbf, Wot, bo, x,
                                                           x1, nullptr, (int)MT, Dd, Dd);
    // LN2
    ln_kernel<<<(int)(MT / 8), blk, 0, stream>>>(x1, g2, b2, h_bf);
    // MLP up + relu (bf16)
    gemm_bf16_kernel<true, true><<<gH, blk, 0, stream>>>(h_bf, W1t, bb1, nullptr,
                                                         nullptr, hidbf, (int)MT, Dh, Dd);
    // MLP down + residual -> out (f32)
    gemm_bf16_kernel<false, false><<<gD, blk, 0, stream>>>(hidbf, W2t, bb2, x1,
                                                           out, nullptr, (int)MT, Dd, Dh);
}
